// BEVLocalizer_9672266351007
// MI455X (gfx1250) — matrix operand order repaired and checked by reading, not compiled, not run on hardware
//
#include <hip/hip_runtime.h>
#include <hip/hip_bf16.h>

// ---------------------------------------------------------------------------
// BEV localizer for MI455X (gfx1250, wave32).
// Pipeline (softmax normalization eliminated — categorical sampling is
// shift-invariant in log space, so only raw relu(q·m)*e^T logits are needed):
//   1) bev_init        : zero the argmax keys
//   2) bev_numvalid    : num_valid per batch
//   3) bev_sample_gemm : WMMA f16 GEMM over the 512 sampled logit rows/batch,
//                        fused Gumbel-max categorical sampling (atomicMax u64)
//   4) bev_poses       : 2-point RANSAC pose construction (512 threads)
//   5) bev_scores      : pose scoring; sim entries recomputed as 32-MAC dots
// ---------------------------------------------------------------------------

#define B_   2
#define N_   1024
#define D_   32
#define I_   128
#define J_   128
#define IJ_  16384
#define K_   256       // poses per batch
#define S_   512       // samples per batch (2K)

typedef __attribute__((ext_vector_type(16))) _Float16 v16h;
typedef __attribute__((ext_vector_type(8)))  float    v8f;

#if __has_builtin(__builtin_amdgcn_logf)
#define FAST_LOG2(x) __builtin_amdgcn_logf(x)   // raw v_log_f32 (log2)
#else
#define FAST_LOG2(x) __log2f(x)
#endif

// --------------------------- small device helpers --------------------------

__device__ __forceinline__ unsigned mix32(unsigned x) {
  x ^= x >> 16; x *= 0x7feb352du;
  x ^= x >> 15; x *= 0x846ca68bu;
  x ^= x >> 16;
  return x;
}

__device__ __forceinline__ int rand_n(int seed, int b, int s) {
  unsigned x = (unsigned)seed * 0x9E3779B9u + (unsigned)(b * S_ + s) * 0x85ebca6bu + 0x1bu;
  return (int)(mix32(x) & (N_ - 1));
}

// Gumbel(0,1) noise. u in (2^-32, 1) => -log2(u) in [~1.7e-10, 32]: never
// denormal, so raw v_log_f32 is exact enough and needs no libm guards.
//   g = -ln(-ln u) = 0.36651292 - ln2 * log2(-log2(u))
__device__ __forceinline__ float gumbelf(unsigned id, int seed) {
  unsigned x = mix32(id * 0x01000193u ^ ((unsigned)seed + 0x811C9DC5u));
  float u  = ((float)x + 0.5f) * 2.3283064365386963e-10f;   // (0,1)
  float l2 = FAST_LOG2(-FAST_LOG2(u));
  return fmaf(l2, -0.69314718055994531f, 0.36651292058166433f);
}

// monotonic float -> uint mapping (total order), result always > 0
__device__ __forceinline__ unsigned ordf(float f) {
  unsigned u = __float_as_uint(f);
  return (u & 0x80000000u) ? ~u : (u | 0x80000000u);
}

// ------------------------------- kernels -----------------------------------

__global__ void bev_init(unsigned long long* __restrict__ keys) {
  int i = blockIdx.x * 256 + threadIdx.x;
  if (i < B_ * S_) keys[i] = 0ull;
}

__global__ __launch_bounds__(256)
void bev_numvalid(const unsigned char* __restrict__ valid_q,
                  float* __restrict__ nvb) {
  const int b = blockIdx.x;
  int cnt = 0;
  for (int n = threadIdx.x; n < N_; n += 256)
    cnt += valid_q[b * N_ + n] ? 1 : 0;
  __shared__ int red[256];
  red[threadIdx.x] = cnt;
  __syncthreads();
  for (int st = 128; st > 0; st >>= 1) {
    if (threadIdx.x < st) red[threadIdx.x] += red[threadIdx.x + st];
    __syncthreads();
  }
  if (threadIdx.x == 0) nvb[b] = fmaxf((float)red[0], 1.0f);
}

// One wave = one 16x16 output tile per j-step; 8 j-steps per wave.
// grid = (B, 32 m-tiles, 16 j-blocks), block = 256 (8 waves).
__global__ __launch_bounds__(256)
void bev_sample_gemm(const float* __restrict__ f_q,
                     const float* __restrict__ f_map,
                     const float* __restrict__ temp,
                     const int*   __restrict__ seedp,
                     unsigned long long* __restrict__ keys) {
  const int b     = blockIdx.x;
  const int mbase = blockIdx.y * 16;          // sample row tile (0..511)
  const int jblk  = blockIdx.z;               // 16 blocks x 1024 cols
  const int wave  = threadIdx.x >> 5;
  const int lane  = threadIdx.x & 31;
  const int half  = lane >> 4;                // 0: lanes 0-15, 1: lanes 16-31
  const int l15   = lane & 15;
  const int seed  = seedp[0];
  const float escale = __expf(temp[0]);

  // ---- A fragment: 16x32 f16. Row = sample row -> hashed query index n.
  // lanes 0-15 hold K {0..15}; lanes 16-31 hold K {16..31}.
  const int srow = mbase + l15;
  const int n    = rand_n(seed, b, srow);
  const float* aq = f_q + ((size_t)(b * N_ + n)) * D_;
  v16h afrag;
#pragma unroll
  for (int h = 0; h < 16; ++h) {
    const int k = h + half * 16;
    afrag[h] = (_Float16)aq[k];
  }

  unsigned long long best[8];
#pragma unroll
  for (int r = 0; r < 8; ++r) best[r] = 0ull;

#pragma unroll 4
  for (int jt = 0; jt < 8; ++jt) {
    const int jbase = jblk * 1024 + jt * 128 + wave * 16;
    const int col   = jbase + l15;           // map cell (flattened i*J+j)
    // ---- B fragment: 32x16 f16. Lane = column; halves split K 0-15/16-31.
    const float* bm = f_map + ((size_t)(b * IJ_ + col)) * D_ + half * 16;
    v16h bfrag;
#pragma unroll
    for (int h = 0; h < 16; ++h) bfrag[h] = (_Float16)bm[h];

    v8f c = {};
    c = __builtin_amdgcn_wmma_f32_16x16x32_f16(
        /*neg_a=*/false, afrag, /*neg_b=*/false, bfrag,
        /*c_mod=*/(short)0, c, /*reuse_a=*/false, /*reuse_b=*/false);

    // ---- fused Gumbel-max epilogue: logit = relu(dot)*e^T + gumbel
#pragma unroll
    for (int r = 0; r < 8; ++r) {
      const int srow_r = mbase + r + half * 8;         // 0..511
      const int ij     = jbase + l15;
      float v = fmaxf(c[r], 0.0f) * escale;
      v += gumbelf(((unsigned)(b * S_ + srow_r) << 14) | (unsigned)ij, seed);
      const unsigned long long key =
          ((unsigned long long)ordf(v) << 32) | (unsigned)ij;
      if (key > best[r]) best[r] = key;
    }
  }

  // ---- reduce the 16 column-lanes of each row (xor <= 8 keeps halves apart)
#pragma unroll
  for (int r = 0; r < 8; ++r) {
    unsigned long long k = best[r];
#pragma unroll
    for (int m = 8; m >= 1; m >>= 1) {
      unsigned long long o = __shfl_xor(k, m, 32);
      if (o > k) k = o;
    }
    best[r] = k;
  }
  if (l15 == 0) {
#pragma unroll
    for (int r = 0; r < 8; ++r) {
      const int srow_r = mbase + r + half * 8;
      atomicMax(&keys[b * S_ + srow_r], best[r]);
    }
  }
}

__global__ __launch_bounds__(256)
void bev_poses(const float* __restrict__ q_xy,
               const int*   __restrict__ seedp,
               const unsigned long long* __restrict__ keys,
               float* __restrict__ pose) {
  const int kk = blockIdx.x * 256 + threadIdx.x;       // 0..511
  if (kk >= B_ * K_) return;
  const int b = kk / K_, k = kk % K_;
  const int seed = seedp[0];
  const int s0 = 2 * k, s1 = 2 * k + 1;
  const int n0 = rand_n(seed, b, s0);
  const int n1 = rand_n(seed, b, s1);
  const int f0 = (int)(keys[b * S_ + s0] & 0xFFFFFFFFull);
  const int f1 = (int)(keys[b * S_ + s1] & 0xFFFFFFFFull);
  const float m0x = ((float)(f0 >> 7)  + 0.5f) * 0.5f;  // mi = flat / J
  const float m0y = ((float)(f0 & 127) + 0.5f) * 0.5f;  // mj = flat % J
  const float m1x = ((float)(f1 >> 7)  + 0.5f) * 0.5f;
  const float m1y = ((float)(f1 & 127) + 0.5f) * 0.5f;
  const float* q0 = q_xy + ((size_t)(b * N_ + n0)) * 2;
  const float* q1 = q_xy + ((size_t)(b * N_ + n1)) * 2;
  const float q0x = q0[0], q0y = q0[1];
  const float q1x = q1[0], q1y = q1[1];
  const float th = atan2f(m1y - m0y, m1x - m0x) - atan2f(q1y - q0y, q1x - q0x);
  const float c = cosf(th), s = sinf(th);
  const float tx = m0x - (c * q0x - s * q0y);
  const float ty = m0y - (s * q0x + c * q0y);
  float* p = pose + (size_t)kk * 4;
  p[0] = c; p[1] = s; p[2] = tx; p[3] = ty;
}

// One block per pose hypothesis; sim entries recomputed as 32-MAC dots
// (f_q + f_map are L2-resident -> no 134 MB sim tensor ever exists).
__global__ __launch_bounds__(256)
void bev_scores(const float* __restrict__ f_q,
                const float* __restrict__ f_map,
                const float* __restrict__ q_xy,
                const float* __restrict__ temp,
                const unsigned char* __restrict__ valid_q,
                const unsigned char* __restrict__ valid_map,
                const float* __restrict__ pose,
                const float* __restrict__ nvb,
                float* __restrict__ out) {
  const int kk = blockIdx.x;                   // 0..511
  const int b  = kk / K_;
  const float c  = pose[kk * 4 + 0];
  const float s  = pose[kk * 4 + 1];
  const float tx = pose[kk * 4 + 2];
  const float ty = pose[kk * 4 + 3];
  const float escale = __expf(temp[0]);

  float acc = 0.0f;
  for (int n = threadIdx.x; n < N_; n += 256) {
    if (!valid_q[b * N_ + n]) continue;
    const float qx = q_xy[((size_t)(b * N_ + n)) * 2 + 0];
    const float qy = q_xy[((size_t)(b * N_ + n)) * 2 + 1];
    const float px = c * qx - s * qy + tx;
    const float py = s * qx + c * qy + ty;
    const int i = (int)floorf(px * 2.0f);      // / CELL (0.5)
    const int j = (int)floorf(py * 2.0f);
    if (i < 0 || i >= I_ || j < 0 || j >= J_) continue;
    const int flat = i * J_ + j;
    if (!valid_map[b * IJ_ + flat]) continue;
    const float4* a4 = (const float4*)(f_q  + ((size_t)(b * N_  + n))   * D_);
    const float4* m4 = (const float4*)(f_map + ((size_t)(b * IJ_ + flat)) * D_);
    float d = 0.0f;
#pragma unroll
    for (int t = 0; t < 8; ++t) {
      const float4 a = a4[t], m = m4[t];
      d = fmaf(a.x, m.x, d); d = fmaf(a.y, m.y, d);
      d = fmaf(a.z, m.z, d); d = fmaf(a.w, m.w, d);
    }
    acc += fmaxf(d, 0.0f) * escale;
  }

  __shared__ float red[256];
  red[threadIdx.x] = acc;
  __syncthreads();
  for (int st = 128; st > 0; st >>= 1) {
    if (threadIdx.x < st) red[threadIdx.x] += red[threadIdx.x + st];
    __syncthreads();
  }
  if (threadIdx.x == 0) out[kk] = red[0] / nvb[b];
}

// ------------------------------ launcher -----------------------------------

extern "C" void kernel_launch(void* const* d_in, const int* in_sizes, int n_in,
                              void* d_out, int out_size, void* d_ws, size_t ws_size,
                              hipStream_t stream) {
  (void)in_sizes; (void)n_in; (void)out_size; (void)ws_size;
  const float* f_q   = (const float*)d_in[0];   // [2,1024,32]
  const float* f_map = (const float*)d_in[1];   // [2,128,128,32]
  const float* q_xy  = (const float*)d_in[2];   // [2,1024,2]
  const float* temp  = (const float*)d_in[3];   // scalar
  const unsigned char* valid_q   = (const unsigned char*)d_in[4];  // [2,1024]
  const unsigned char* valid_map = (const unsigned char*)d_in[5];  // [2,128,128]
  const int* seed = (const int*)d_in[6];        // scalar
  float* out = (float*)d_out;                   // [2,256]

  char* ws = (char*)d_ws;                       // ~16 KB total
  unsigned long long* keys = (unsigned long long*)(ws);          // 1024 u64
  float* pose = (float*)(ws + 8192);                             // 512 * 4 f32
  float* nvb  = (float*)(ws + 16384);                            // 2 f32

  bev_init<<<4, 256, 0, stream>>>(keys);
  bev_numvalid<<<B_, 256, 0, stream>>>(valid_q, nvb);
  dim3 g(B_, S_ / 16, 16);
  bev_sample_gemm<<<g, 256, 0, stream>>>(f_q, f_map, temp, seed, keys);
  bev_poses<<<B_, 256, 0, stream>>>(q_xy, seed, keys, pose);
  bev_scores<<<B_ * K_, 256, 0, stream>>>(f_q, f_map, q_xy, temp, valid_q,
                                          valid_map, pose, nvb, out);
}
